// TemporalAttention_15822659519214
// MI455X (gfx1250) — matrix operand order repaired and checked by reading, not compiled, not run on hardware
//
#include <hip/hip_runtime.h>

// Problem sizes (compile-time constants from the reference)
#define B_  64
#define C_  256
#define T_  4096
#define H_  128
#define TT  128    // T-tile per block in the scores kernel
#define XP  40     // padded bf16 row stride for the LDS x-slice ([t][k])

typedef __attribute__((ext_vector_type(16))) __bf16 v16bf;
typedef __attribute__((ext_vector_type(8)))  float  v8f;

struct US16 { unsigned short u[16]; };   // 32 bytes, bit-cast target for v16bf
struct U4x2 { uint4 a, b; };             // 32 bytes

// fp32 -> bf16 round-to-nearest-even
__device__ __forceinline__ unsigned short f2bf(float f) {
  unsigned int x = __builtin_bit_cast(unsigned int, f);
  unsigned int r = 0x7FFFu + ((x >> 16) & 1u);
  return (unsigned short)((x + r) >> 16);
}

// B-fragment: 16 contiguous bf16 from LDS (two 16B chunks -> ds_load_b128 x2)
__device__ __forceinline__ v16bf load_b_frag(const unsigned short* p) {
  U4x2 t;
  t.a = *(const uint4*)(p);
  t.b = *(const uint4*)(p + 8);
  return __builtin_bit_cast(v16bf, t);
}

// ---------------------------------------------------------------------------
// Kernel 1: scores[b,t] = W2 . relu(W1^T x[b,:,t] + b1) + b2
// Grid: (T/TT, B), block 256 threads = 8 waves. Wave w owns H rows 16w..16w+15.
// ---------------------------------------------------------------------------
__global__ __launch_bounds__(256)
void ta_scores_kernel(const float* __restrict__ x, const float* __restrict__ W1,
                      const float* __restrict__ b1, const float* __restrict__ W2,
                      const float* __restrict__ b2, float* __restrict__ scores)
{
  __shared__ __align__(16) unsigned short xs[TT * XP];  // x slice, [t][k] bf16
  __shared__ float sacc[TT];                            // per-column score accum

  const int tid  = threadIdx.x;
  const int lane = tid & 31;
  const int wave = tid >> 5;
  const int b    = blockIdx.y;
  const int t0   = blockIdx.x * TT;

  // --- Preload this wave's A fragments (W1^T slice) into registers as bf16.
  // A k order used here (contiguous): lane<16 -> K in {0..15}, lane>=16 ->
  // K in {16..31}; element i = K base + i (matches B's contiguous order).
  v16bf aW[8];
  {
    const int m     = wave * 16 + (lane & 15);  // A row = W1 column
    const int khalf = lane >> 4;
#pragma unroll
    for (int ks = 0; ks < 8; ++ks) {
      US16 t;
#pragma unroll
      for (int j = 0; j < 8; ++j) {
        const int k0 = ks * 32 + khalf * 16 + j;
        t.u[j]     = f2bf(W1[(size_t)k0        * H_ + m]);   // K = base + j
        t.u[j + 8] = f2bf(W1[(size_t)(k0 + 8) * H_ + m]);   // K = base + 8 + j
      }
      aW[ks] = __builtin_bit_cast(v16bf, t);
    }
  }

  if (tid < TT) sacc[tid] = 0.0f;

  v8f acc[8] = {};
  const float* xb = x + (size_t)b * C_ * T_ + t0;

  for (int ks = 0; ks < 8; ++ks) {          // K steps over C=256, 32 at a time
    __syncthreads();                        // xs reuse guard (and sacc init @ks=0)
    // Stage 32(K) x 128(T) slice of x into LDS, transposed to [t][k] bf16.
    for (int i = tid; i < 32 * (TT / 4); i += 256) {
      const int kc = i >> 5;                // 0..31  (TT/4 float4 per K-row)
      const int tq = i & 31;
      float4 v = *(const float4*)(xb + (size_t)(ks * 32 + kc) * T_ + tq * 4);
      const int t = tq * 4;
      xs[(t + 0) * XP + kc] = f2bf(v.x);
      xs[(t + 1) * XP + kc] = f2bf(v.y);
      xs[(t + 2) * XP + kc] = f2bf(v.z);
      xs[(t + 3) * XP + kc] = f2bf(v.w);
    }
    __syncthreads();

    // B k order used here (contiguous): lane<16 -> col=lane, K=0..15; lane>=16 ->
    // col=lane-16, K=16..31. With xs[t][k] those 16 K are contiguous.
    const int bn = lane & 15;
    const int bk = (lane >> 4) * 16;
#pragma unroll
    for (int nt = 0; nt < 8; ++nt) {
      v16bf bfrag = load_b_frag(&xs[(nt * 16 + bn) * XP + bk]);
      acc[nt] = __builtin_amdgcn_wmma_f32_16x16x32_bf16(
          false, aW[ks], false, bfrag, (short)0, acc[nt], false, false);
    }
  }

  // Epilogue: C/D layout -> lane 0-15: M=r, N=lane; lane 16-31: M=r+8, N=lane-16.
  const int n     = lane & 15;
  const int mbase = wave * 16 + (lane >> 4) * 8;
  float bias[8], wv[8];
#pragma unroll
  for (int r = 0; r < 8; ++r) {
    bias[r] = b1[mbase + r];
    wv[r]   = W2[mbase + r];
  }
#pragma unroll
  for (int nt = 0; nt < 8; ++nt) {
    float partial = 0.0f;
#pragma unroll
    for (int r = 0; r < 8; ++r) {
      float h = acc[nt][r] + bias[r];
      h = h > 0.0f ? h : 0.0f;
      partial += wv[r] * h;                // fold W2 immediately (post-ReLU)
    }
    atomicAdd(&sacc[nt * 16 + n], partial);  // ds_add_f32
  }
  __syncthreads();
  if (tid < TT) scores[(size_t)b * T_ + t0 + tid] = sacc[tid] + b2[0];
}

// ---------------------------------------------------------------------------
// Kernel 2: softmax over T per batch. One block per b; row lives in LDS.
// ---------------------------------------------------------------------------
__global__ __launch_bounds__(256)
void ta_softmax_kernel(const float* __restrict__ scores, float* __restrict__ weights)
{
  __shared__ float s[T_];
  __shared__ float red[8];
  const int b = blockIdx.x, tid = threadIdx.x;

  float mx = -__builtin_inff();
  for (int i = tid; i < T_; i += 256) {
    float v = scores[(size_t)b * T_ + i];
    s[i] = v;
    mx = fmaxf(mx, v);
  }
#pragma unroll
  for (int o = 16; o > 0; o >>= 1) mx = fmaxf(mx, __shfl_xor(mx, o));
  if ((tid & 31) == 0) red[tid >> 5] = mx;
  __syncthreads();
  if (tid == 0) {
    float m = red[0];
    for (int i = 1; i < 8; ++i) m = fmaxf(m, red[i]);
    red[0] = m;
  }
  __syncthreads();
  mx = red[0];

  float sum = 0.0f;
  for (int i = tid; i < T_; i += 256) {
    float e = __expf(s[i] - mx);
    s[i] = e;
    sum += e;
  }
#pragma unroll
  for (int o = 16; o > 0; o >>= 1) sum += __shfl_xor(sum, o);
  __syncthreads();                       // done reading red[0] as max
  if ((tid & 31) == 0) red[tid >> 5] = sum;
  __syncthreads();
  if (tid == 0) {
    float t = 0.0f;
    for (int i = 0; i < 8; ++i) t += red[i];
    red[0] = t;
  }
  __syncthreads();
  const float inv = 1.0f / red[0];
  for (int i = tid; i < T_; i += 256) weights[(size_t)b * T_ + i] = s[i] * inv;
}

// ---------------------------------------------------------------------------
// Kernel 3: out[b,c] = sum_t x[b,c,t] * w[b,t]. One block per (c,b).
// Pure bandwidth: coalesced float4 reads of x; w row is L2-hot.
// ---------------------------------------------------------------------------
__global__ __launch_bounds__(256)
void ta_wsum_kernel(const float* __restrict__ x, const float* __restrict__ weights,
                    float* __restrict__ out)
{
  const int c = blockIdx.x, b = blockIdx.y, tid = threadIdx.x;
  const float4* x4 = (const float4*)(x + ((size_t)b * C_ + c) * T_);
  const float4* w4 = (const float4*)(weights + (size_t)b * T_);

  float acc = 0.0f;
  for (int i = tid; i < T_ / 4; i += 256) {
    float4 a = x4[i], w = w4[i];
    acc += a.x * w.x + a.y * w.y + a.z * w.z + a.w * w.w;
  }
#pragma unroll
  for (int o = 16; o > 0; o >>= 1) acc += __shfl_xor(acc, o);

  __shared__ float red[8];
  if ((tid & 31) == 0) red[tid >> 5] = acc;
  __syncthreads();
  if (tid == 0) {
    float s = 0.0f;
    for (int i = 0; i < 8; ++i) s += red[i];
    out[(size_t)b * C_ + c] = s;
  }
}

// ---------------------------------------------------------------------------
extern "C" void kernel_launch(void* const* d_in, const int* in_sizes, int n_in,
                              void* d_out, int out_size, void* d_ws, size_t ws_size,
                              hipStream_t stream) {
  const float* x  = (const float*)d_in[0];   // [B, C, T]
  const float* W1 = (const float*)d_in[1];   // [C, H]
  const float* b1 = (const float*)d_in[2];   // [H]
  const float* W2 = (const float*)d_in[3];   // [H, 1]
  const float* b2 = (const float*)d_in[4];   // [1]
  float* out = (float*)d_out;                // [B, C]

  float* scores  = (float*)d_ws;             // [B, T]  (1 MB)
  float* weights = scores + (size_t)B_ * T_; // [B, T]  (1 MB)

  ta_scores_kernel<<<dim3(T_ / TT, B_), 256, 0, stream>>>(x, W1, b1, W2, b2, scores);
  ta_softmax_kernel<<<dim3(B_), 256, 0, stream>>>(scores, weights);
  ta_wsum_kernel<<<dim3(C_, B_), 256, 0, stream>>>(x, weights, out);
}
